// ST_Fusion_41807211659348
// MI455X (gfx1250) — compile-verified, matrix operand order checked by reading, not run on hardware
//
#include <hip/hip_runtime.h>
#include <math.h>

// ---------------------------------------------------------------------------
// Types for CDNA5 WMMA
// ---------------------------------------------------------------------------
typedef __attribute__((ext_vector_type(16))) __bf16 v16bf;
typedef __attribute__((ext_vector_type(8)))  float  v8f;

#define NFc 64
#define NFRAMESc 5
#define CENTERc 2
#define DGc 8
#define BBc 8
#define HHc 96
#define WWc 96

// ---------------------------------------------------------------------------
// Helpers
// ---------------------------------------------------------------------------
__device__ __forceinline__ unsigned short f2bf(float f) {
    union { float f; unsigned int u; } v; v.f = f;
    unsigned int u = v.u;
    u += 0x7fffu + ((u >> 16) & 1u);     // round-to-nearest-even
    return (unsigned short)(u >> 16);
}
__device__ __forceinline__ float cvtIn(float x) { return x; }
__device__ __forceinline__ float cvtIn(unsigned short x) {
    union { unsigned int u; float f; } v; v.u = ((unsigned int)x) << 16; return v.f;
}
// K-index of element e (0..15) of a 16x16x32 16-bit A/B fragment for `lane`
// per CDNA5 ISA 7.12.2 (B uses the mirrored convention with n = lane & 15).
__device__ __forceinline__ int kmap(int lane, int e) {
    int half = (lane >> 4) & 1;
    int g = e >> 1, p = e & 1;
    int base = (g < 4) ? (half * 8 + g * 2) : (16 + half * 8 + (g - 4) * 2);
    return base + p;
}
__device__ __forceinline__ float actf(float v, int act) {
    if (act == 1) return v >= 0.f ? v : 0.1f * v;   // leaky relu
    if (act == 2) return v >= 0.f ? v : 0.f;        // relu
    return v;
}
__device__ __forceinline__ float sigm(float x) { return 1.f / (1.f + __expf(-x)); }

// ---------------------------------------------------------------------------
// Weight packing: fp32 OIHW -> bf16 in WMMA A-fragment order
// layout: [mt][tap][q][lane][e]  (512 halves per (mt,tap,q) -> one frag/lane)
// ---------------------------------------------------------------------------
__global__ void pack_kernel(const float* __restrict__ w, unsigned short* __restrict__ wpk,
                            int Cout, int Cin, int taps, int total) {
    int idx = blockIdx.x * blockDim.x + threadIdx.x;
    if (idx >= total) return;
    int e = idx & 15, lane = (idx >> 4) & 31;
    int r = idx >> 9;
    int qn = (Cin + 31) >> 5;
    int q = r % qn; r /= qn;
    int t = r % taps; int mt = r / taps;
    int co = mt * 16 + (lane & 15);
    int ci = q * 32 + kmap(lane, e);
    float v = 0.f;
    if (co < Cout && ci < Cin) v = w[((size_t)co * Cin + ci) * taps + t];
    wpk[idx] = f2bf(v);
}

// ---------------------------------------------------------------------------
// Implicit-GEMM conv (ks in {1,3}) with bf16 WMMA, fp32 accumulate.
// Block = 128 threads (4 waves). Wave `mt` computes a 16(Cout)x32(pixel) tile:
// two 16x16 WMMA tiles sharing the LDS-staged B operand across all 4 waves.
// `active` is forced into an SGPR (readfirstlane) so the WMMA region is
// guarded by a scalar branch and EXEC remains all-ones (ISA 7.12 requirement).
// ---------------------------------------------------------------------------
template <typename Tin>
__global__ void __launch_bounds__(128)
wconv_kernel(const Tin* __restrict__ in, int inCtot, int inCoff, int Cin,
             const unsigned short* __restrict__ wpk, const float* __restrict__ bias,
             float* __restrict__ out, int outCtot, int outCoff, int Cout,
             const float* __restrict__ res, int resCtot, int resCoff,
             int Hh, int Ww, int ks, int pad, int act) {
    __shared__ __align__(32) unsigned int lbuf[512];   // two 32x16 bf16 B tiles
    const int tid  = threadIdx.x;
    const int lane = tid & 31, wave = tid >> 5;
    const int wTiles = (Ww + 31) >> 5;
    const int ty = blockIdx.x / wTiles;
    const int x0 = (blockIdx.x % wTiles) << 5;
    const int b  = blockIdx.y;
    const int mt = __builtin_amdgcn_readfirstlane(blockIdx.z * 4 + wave);  // wave-uniform SGPR
    const int mtT = (Cout + 15) >> 4;
    const bool active = mt < mtT;                      // scalar condition
    const int qn = (Cin + 31) >> 5;
    const int taps = ks * ks;
    const size_t plane = (size_t)Hh * Ww;
    const Tin* inB = in + ((size_t)b * inCtot + inCoff) * plane;

    v8f acc0 = {}; v8f acc1 = {};
    for (int t = 0; t < taps; ++t) {
        const int dy = t / ks - pad, dx = t % ks - pad;
        const int y = ty + dy;
        const bool yok = (y >= 0) && (y < Hh);
        for (int q = 0; q < qn; ++q) {
            // stage B tiles into LDS in fragment layout (2 dwords per thread x 2)
            for (int d = tid; d < 512; d += 128) {
                const int lane_l = (d >> 3) & 31;
                const int g = d & 7;
                const int tile = d >> 8;
                const int n = (tile << 4) + (lane_l & 15);
                const int x = x0 + n + dx;
                unsigned int pk = 0u;
                if (yok && x >= 0 && x < Ww) {
                    const int k0 = q * 32 + kmap(lane_l, 2 * g);
                    const int k1 = q * 32 + kmap(lane_l, 2 * g + 1);
                    float v0 = (k0 < Cin) ? cvtIn(inB[(size_t)k0 * plane + (size_t)y * Ww + x]) : 0.f;
                    float v1 = (k1 < Cin) ? cvtIn(inB[(size_t)k1 * plane + (size_t)y * Ww + x]) : 0.f;
                    pk = (unsigned int)f2bf(v0) | ((unsigned int)f2bf(v1) << 16);
                }
                lbuf[d] = pk;
            }
            __syncthreads();
            if (active) {
                const size_t aoff = ((((size_t)mt * taps + t) * qn + q) << 9) + ((size_t)lane << 4);
                v16bf a  = *reinterpret_cast<const v16bf*>(wpk + aoff);
                __builtin_prefetch(wpk + aoff + 512, 0, 1);   // next fragment -> global_prefetch
                v16bf b0 = *reinterpret_cast<const v16bf*>(&lbuf[(size_t)lane * 8]);
                v16bf b1 = *reinterpret_cast<const v16bf*>(&lbuf[256 + (size_t)lane * 8]);
                acc0 = __builtin_amdgcn_wmma_f32_16x16x32_bf16(false, a, false, b0, (short)0, acc0, false, false);
                acc1 = __builtin_amdgcn_wmma_f32_16x16x32_bf16(false, a, false, b1, (short)0, acc1, false, false);
            }
            __syncthreads();
        }
    }
    if (!active) return;
    const int half = lane >> 4, l15 = lane & 15;
    const size_t outBase = ((size_t)b * outCtot + outCoff) * plane;
#pragma unroll
    for (int r = 0; r < 8; ++r) {
        const int co = mt * 16 + r + half * 8;      // C/D layout per ISA 7.12.2
        if (co >= Cout) continue;
        const float bv = bias ? bias[co] : 0.f;
        int px = x0 + l15;
        if (px < Ww) {
            float v = acc0[r] + bv;
            if (res) v += res[((size_t)b * resCtot + resCoff + co) * plane + (size_t)ty * Ww + px];
            out[outBase + (size_t)co * plane + (size_t)ty * Ww + px] = actf(v, act);
        }
        px = x0 + 16 + l15;
        if (px < Ww) {
            float v = acc1[r] + bv;
            if (res) v += res[((size_t)b * resCtot + resCoff + co) * plane + (size_t)ty * Ww + px];
            out[outBase + (size_t)co * plane + (size_t)ty * Ww + px] = actf(v, act);
        }
    }
}

// ---------------------------------------------------------------------------
// Elementwise / glue kernels
// ---------------------------------------------------------------------------
__global__ void copy_slice_kernel(const float* __restrict__ in, int inCtot, int inCoff,
                                  float* __restrict__ out, int outCtot, int outCoff,
                                  int C, int plane, int total) {
    int idx = blockIdx.x * blockDim.x + threadIdx.x;
    if (idx >= total) return;
    int p = idx % plane; int r = idx / plane;
    int c = r % C; int b = r / C;
    out[((size_t)b * outCtot + outCoff + c) * plane + p] =
        in[((size_t)b * inCtot + inCoff + c) * plane + p];
}

// DCNv2 bilinear sampling: build modulated `val` (bf16) laid out [b][(d*8+c)*9+k][plane]
__global__ void dcn_sample_kernel(const float* __restrict__ fea, int frame,
                                  const float* __restrict__ om,
                                  unsigned short* __restrict__ val, int total) {
    int idx = blockIdx.x * blockDim.x + threadIdx.x;
    if (idx >= total) return;
    const int plane = HHc * WWc;
    int p = idx % plane; int r = idx / plane;
    int k = r % 9; r /= 9;
    int d = r % DGc; int b = r / DGc;
    int h = p / WWc, w = p % WWc;
    size_t omb = ((size_t)b * 216 + d * 27) * plane + p;
    float dy = om[omb + (size_t)k * plane];
    float dx = om[omb + (size_t)(9 + k) * plane];
    float mm = sigm(om[omb + (size_t)(18 + k) * plane]);
    float py = (float)h + (float)(k / 3 - 1) + dy;
    float px = (float)w + (float)(k % 3 - 1) + dx;
    float y0f = floorf(py), x0f = floorf(px);
    float ay = py - y0f, ax = px - x0f;
    int iy0 = (int)y0f, ix0 = (int)x0f;
    int iy1 = iy0 + 1,  ix1 = ix0 + 1;
    float vy0 = (iy0 >= 0 && iy0 < HHc) ? 1.f : 0.f;
    float vy1 = (iy1 >= 0 && iy1 < HHc) ? 1.f : 0.f;
    float vx0 = (ix0 >= 0 && ix0 < WWc) ? 1.f : 0.f;
    float vx1 = (ix1 >= 0 && ix1 < WWc) ? 1.f : 0.f;
    int cy0 = min(max(iy0, 0), HHc - 1), cy1 = min(max(iy1, 0), HHc - 1);
    int cx0 = min(max(ix0, 0), WWc - 1), cx1 = min(max(ix1, 0), WWc - 1);
    float w00 = (1.f - ay) * (1.f - ax) * vy0 * vx0;
    float w01 = (1.f - ay) * ax * vy0 * vx1;
    float w10 = ay * (1.f - ax) * vy1 * vx0;
    float w11 = ay * ax * vy1 * vx1;
    const float* xb = fea + ((size_t)(b * NFRAMESc + frame) * NFc + d * 8) * plane;
    size_t vb = ((size_t)b * 576 + (size_t)(d * 8) * 9 + k) * plane + p;
#pragma unroll
    for (int c = 0; c < 8; ++c) {
        const float* xc = xb + (size_t)c * plane;
        float v = w00 * xc[cy0 * WWc + cx0] + w01 * xc[cy0 * WWc + cx1] +
                  w10 * xc[cy1 * WWc + cx0] + w11 * xc[cy1 * WWc + cx1];
        val[vb + (size_t)c * 9 * plane] = f2bf(v * mm);
    }
}

__global__ void corr_kernel(const float* __restrict__ embN, const float* __restrict__ embR,
                            float* __restrict__ cor, int frame, int total) {
    int idx = blockIdx.x * blockDim.x + threadIdx.x;
    if (idx >= total) return;
    const int plane = HHc * WWc;
    int p = idx % plane; int b = idx / plane;
    float s = 0.f;
    for (int c = 0; c < NFc; ++c)
        s += embN[((size_t)b * NFc + c) * plane + p] * embR[((size_t)b * NFc + c) * plane + p];
    cor[((size_t)b * NFRAMESc + frame) * plane + p] = s;
}

__global__ void tam_kernel(const float* __restrict__ fea, const float* __restrict__ cor,
                           float* __restrict__ cat1, int total) {
    int idx = blockIdx.x * blockDim.x + threadIdx.x;
    if (idx >= total) return;
    const int plane = HHc * WWc;
    int p = idx % plane; int r = idx / plane;
    int ch = r % (NFRAMESc * NFc); int b = r / (NFRAMESc * NFc);
    float s = sigm(cor[((size_t)b * NFRAMESc + (ch >> 6)) * plane + p]);
    cat1[((size_t)b * 384 + ch) * plane + p] = fea[((size_t)b * 320 + ch) * plane + p] * s;
}

__global__ void pool_kernel(const float* __restrict__ in, int inCtot, int inCoff,
                            float* __restrict__ out, int outCtot, int outCoff,
                            int C, int Hi, int Wi, int ismax, int total) {
    int idx = blockIdx.x * blockDim.x + threadIdx.x;
    if (idx >= total) return;
    int Ho = Hi >> 1, Wo = Wi >> 1;
    int wo = idx % Wo; int r = idx / Wo;
    int ho = r % Ho; r /= Ho;
    int c = r % C; int b = r / C;
    const float* ip = in + ((size_t)b * inCtot + inCoff + c) * Hi * Wi;
    float m = -3.402823e38f, s = 0.f;
    for (int jy = 0; jy < 3; ++jy) {
        int y = 2 * ho + jy - 1;
        if (y < 0 || y >= Hi) continue;
        for (int jx = 0; jx < 3; ++jx) {
            int x = 2 * wo + jx - 1;
            if (x < 0 || x >= Wi) continue;
            float v = ip[y * Wi + x];
            m = fmaxf(m, v); s += v;
        }
    }
    out[((size_t)b * outCtot + outCoff + c) * Ho * Wo + ho * Wo + wo] = ismax ? m : s * (1.f / 9.f);
}

__global__ void up2_kernel(const float* __restrict__ in, const float* __restrict__ addsrc,
                           float* __restrict__ out, int C, int Hi, int Wi, int total) {
    int idx = blockIdx.x * blockDim.x + threadIdx.x;
    if (idx >= total) return;
    int Ho = Hi * 2, Wo = Wi * 2;
    int wo = idx % Wo; int r = idx / Wo;
    int ho = r % Ho; r /= Ho;
    int c = r % C; int b = r / C;
    float sy = (ho + 0.5f) * 0.5f - 0.5f;
    float sx = (wo + 0.5f) * 0.5f - 0.5f;
    float y0f = floorf(sy), x0f = floorf(sx);
    float fy = sy - y0f, fx = sx - x0f;
    int y0 = (int)y0f, x0 = (int)x0f;
    int y0c = min(max(y0, 0), Hi - 1), y1c = min(max(y0 + 1, 0), Hi - 1);
    int x0c = min(max(x0, 0), Wi - 1), x1c = min(max(x0 + 1, 0), Wi - 1);
    const float* ip = in + ((size_t)b * C + c) * Hi * Wi;
    float v = (1.f - fy) * ((1.f - fx) * ip[y0c * Wi + x0c] + fx * ip[y0c * Wi + x1c]) +
              fy * ((1.f - fx) * ip[y1c * Wi + x0c] + fx * ip[y1c * Wi + x1c]);
    size_t o = ((size_t)b * C + c) * Ho * Wo + (size_t)ho * Wo + wo;
    if (addsrc) v += addsrc[o];
    out[o] = v;
}

__global__ void final_kernel(const float* __restrict__ fea_o, const float* __restrict__ att,
                             const float* __restrict__ add, float* __restrict__ out, int total) {
    int idx = blockIdx.x * blockDim.x + threadIdx.x;
    if (idx >= total) return;
    out[idx] = fea_o[idx] * sigm(att[idx]) * 2.f + add[idx];
}

// ---------------------------------------------------------------------------
// Host orchestration
// ---------------------------------------------------------------------------
enum Lyr { TATT1 = 0, TATT2, OFF1, OFF2, DCNOFF, DCN, RES1, RES2, FFTAM, FF,
           SA1, SA2, SA3, SA4, SA5, SL1, SL2, SL3, ADD1, ADD2 };

static const int LC[20][3] = {
    {64, 64, 3},  {64, 64, 3},  {64, 128, 3}, {64, 64, 3},  {216, 64, 3},
    {64, 576, 1}, {64, 64, 3},  {64, 64, 3},  {64, 384, 1}, {64, 384, 1},
    {64, 384, 1}, {64, 128, 1}, {64, 64, 3},  {64, 64, 1},  {64, 64, 3},
    {64, 64, 1},  {64, 128, 3}, {64, 64, 3},  {64, 64, 1},  {64, 64, 1}};

extern "C" void kernel_launch(void* const* d_in, const int* in_sizes, int n_in,
                              void* d_out, int out_size, void* d_ws, size_t ws_size,
                              hipStream_t stream) {
    (void)in_sizes; (void)n_in; (void)out_size; (void)ws_size;
    const float* fea = (const float*)d_in[0];
    auto Wt = [&](int l) { return (const float*)d_in[1 + 2 * l]; };
    auto Bs = [&](int l) { return (const float*)d_in[2 + 2 * l]; };

    size_t cur = 0;
    auto alloc = [&](size_t bytes) -> char* {
        cur = (cur + 255) & ~(size_t)255;
        char* p = (char*)d_ws + cur;
        cur += bytes;
        return p;
    };
    auto g1 = [](int n) { return dim3((unsigned)((n + 255) / 256)); };

    // ---- pack all weights to bf16 WMMA fragments ----
    unsigned short* wpk[20];
    for (int l = 0; l < 20; ++l) {
        int cout = LC[l][0], cin = LC[l][1], ks = LC[l][2], taps = ks * ks;
        int mtT = (cout + 15) / 16, qn = (cin + 31) / 32;
        int total = mtT * taps * qn * 512;
        wpk[l] = (unsigned short*)alloc((size_t)total * 2);
        pack_kernel<<<g1(total), 256, 0, stream>>>(Wt(l), wpk[l], cout, cin, taps, total);
    }

    const size_t P96 = (size_t)HHc * WWc, P48 = 48 * 48, P24 = 24 * 24;
    float* emb_ref = (float*)alloc(64ull * BBc * P96 * 4);
    float* emb_nbr = (float*)alloc(64ull * BBc * P96 * 4);
    float* catnr   = (float*)alloc(128ull * BBc * P96 * 4);
    float* off_a   = (float*)alloc(64ull * BBc * P96 * 4);
    float* off_b   = (float*)alloc(64ull * BBc * P96 * 4);
    float* om      = (float*)alloc(216ull * BBc * P96 * 4);
    unsigned short* val = (unsigned short*)alloc(576ull * BBc * P96 * 2);
    float* cor     = (float*)alloc((size_t)NFRAMESc * BBc * P96 * 4);
    float* cat1    = (float*)alloc(384ull * BBc * P96 * 4);
    float* cat2    = (float*)alloc(384ull * BBc * P96 * 4);
    float* rbt     = (float*)alloc(64ull * BBc * P96 * 4);
    float* fea_o   = (float*)alloc(64ull * BBc * P96 * 4);
    float* att96   = (float*)alloc(64ull * BBc * P96 * 4);
    float* attp48  = (float*)alloc(128ull * BBc * P48 * 4);
    float* att48   = (float*)alloc(64ull * BBc * P48 * 4);
    float* attL48  = (float*)alloc(64ull * BBc * P48 * 4);
    float* attLp24 = (float*)alloc(128ull * BBc * P24 * 4);
    float* attL24  = (float*)alloc(64ull * BBc * P24 * 4);
    float* attL24b = (float*)alloc(64ull * BBc * P24 * 4);
    float* attA48  = (float*)alloc(64ull * BBc * P48 * 4);
    float* att48b  = (float*)alloc(64ull * BBc * P48 * 4);
    float* att48c  = (float*)alloc(64ull * BBc * P48 * 4);
    float* att96b  = (float*)alloc(64ull * BBc * P96 * 4);
    float* att96c  = (float*)alloc(64ull * BBc * P96 * 4);
    float* addt    = (float*)alloc(64ull * BBc * P96 * 4);
    float* attadd  = (float*)alloc(64ull * BBc * P96 * 4);

    // generic conv launcher
    auto conv = [&](const void* in, bool bf16in, int inCtot, int inCoff, int Cin, int l,
                    float* out, int outCtot, int outCoff,
                    const float* res, int resCtot, int resCoff,
                    int H, int Wd, int act) {
        int cout = LC[l][0], ks = LC[l][2], pad = (ks == 3) ? 1 : 0;
        dim3 grid((unsigned)(H * ((Wd + 31) / 32)), BBc, (unsigned)((cout + 63) / 64));
        if (bf16in)
            wconv_kernel<unsigned short><<<grid, 128, 0, stream>>>(
                (const unsigned short*)in, inCtot, inCoff, Cin, wpk[l], Bs(l),
                out, outCtot, outCoff, cout, res, resCtot, resCoff, H, Wd, ks, pad, act);
        else
            wconv_kernel<float><<<grid, 128, 0, stream>>>(
                (const float*)in, inCtot, inCoff, Cin, wpk[l], Bs(l),
                out, outCtot, outCoff, cout, res, resCtot, resCoff, H, Wd, ks, pad, act);
    };

    const int n64 = (int)(64ull * BBc * P96);

    // emb_ref = conv(tAtt_2, ref)
    conv(fea, false, 320, CENTERc * 64, 64, TATT2, emb_ref, 64, 0, nullptr, 0, 0, 96, 96, 0);

    // per-frame alignment
    for (int i = 0; i < NFRAMESc; ++i) {
        copy_slice_kernel<<<g1(n64), 256, 0, stream>>>(fea, 320, i * 64, catnr, 128, 0, 64, (int)P96, n64);
        copy_slice_kernel<<<g1(n64), 256, 0, stream>>>(fea, 320, CENTERc * 64, catnr, 128, 64, 64, (int)P96, n64);
        conv(catnr, false, 128, 0, 128, OFF1, off_a, 64, 0, nullptr, 0, 0, 96, 96, 1);
        conv(off_a, false, 64, 0, 64, OFF2, off_b, 64, 0, nullptr, 0, 0, 96, 96, 1);
        conv(off_b, false, 64, 0, 64, DCNOFF, om, 216, 0, nullptr, 0, 0, 96, 96, 0);
        int nsamp = (int)((size_t)BBc * DGc * 9 * P96);
        dcn_sample_kernel<<<g1(nsamp), 256, 0, stream>>>(fea, i, om, val, nsamp);
        // dcn as 64x576 GEMM over pixels, fused lrelu, write into cat2 aligned slice
        conv(val, true, 576, 0, 576, DCN, cat2, 384, 64 + i * 64, nullptr, 0, 0, 96, 96, 1);
        conv(cat2, false, 384, 64 + i * 64, 64, TATT1, emb_nbr, 64, 0, nullptr, 0, 0, 96, 96, 0);
        int ncor = (int)((size_t)BBc * P96);
        corr_kernel<<<g1(ncor), 256, 0, stream>>>(emb_nbr, emb_ref, cor, i, ncor);
    }

    // temporal attention modulate -> cat1[0:320]
    int ntam = (int)((size_t)BBc * 320 * P96);
    tam_kernel<<<g1(ntam), 256, 0, stream>>>(fea, cor, cat1, ntam);
    // ref residual block -> cat1[320:384]
    conv(fea, false, 320, CENTERc * 64, 64, RES1, rbt, 64, 0, nullptr, 0, 0, 96, 96, 2);
    conv(rbt, false, 64, 0, 64, RES2, cat1, 384, 320, fea, 320, CENTERc * 64, 96, 96, 0);
    // fusion
    conv(cat1, false, 384, 0, 384, FFTAM, cat2, 384, 0, nullptr, 0, 0, 96, 96, 1);
    conv(cat2, false, 384, 0, 384, FF, fea_o, 64, 0, nullptr, 0, 0, 96, 96, 1);
    // spatial attention pyramid
    conv(cat2, false, 384, 0, 384, SA1, att96, 64, 0, nullptr, 0, 0, 96, 96, 1);
    int np48 = (int)(64ull * BBc * P48);
    pool_kernel<<<g1(np48), 256, 0, stream>>>(att96, 64, 0, attp48, 128, 0, 64, 96, 96, 1, np48);
    pool_kernel<<<g1(np48), 256, 0, stream>>>(att96, 64, 0, attp48, 128, 64, 64, 96, 96, 0, np48);
    conv(attp48, false, 128, 0, 128, SA2, att48, 64, 0, nullptr, 0, 0, 48, 48, 1);
    conv(att48, false, 64, 0, 64, SL1, attL48, 64, 0, nullptr, 0, 0, 48, 48, 1);
    int np24 = (int)(64ull * BBc * P24);
    pool_kernel<<<g1(np24), 256, 0, stream>>>(attL48, 64, 0, attLp24, 128, 0, 64, 48, 48, 1, np24);
    pool_kernel<<<g1(np24), 256, 0, stream>>>(attL48, 64, 0, attLp24, 128, 64, 64, 48, 48, 0, np24);
    conv(attLp24, false, 128, 0, 128, SL2, attL24, 64, 0, nullptr, 0, 0, 24, 24, 1);
    conv(attL24, false, 64, 0, 64, SL3, attL24b, 64, 0, nullptr, 0, 0, 24, 24, 1);
    conv(att48, false, 64, 0, 64, SA3, attA48, 64, 0, nullptr, 0, 0, 48, 48, 1);
    up2_kernel<<<g1(np48), 256, 0, stream>>>(attL24b, attA48, att48b, 64, 24, 24, np48);
    conv(att48b, false, 64, 0, 64, SA4, att48c, 64, 0, nullptr, 0, 0, 48, 48, 1);
    up2_kernel<<<g1(n64), 256, 0, stream>>>(att48c, nullptr, att96b, 64, 48, 48, n64);
    conv(att96b, false, 64, 0, 64, SA5, att96c, 64, 0, nullptr, 0, 0, 96, 96, 0);
    conv(att96c, false, 64, 0, 64, ADD1, addt, 64, 0, nullptr, 0, 0, 96, 96, 1);
    conv(addt, false, 64, 0, 64, ADD2, attadd, 64, 0, nullptr, 0, 0, 96, 96, 0);
    final_kernel<<<g1(n64), 256, 0, stream>>>(fea_o, att96c, attadd, (float*)d_out, n64);
}
